// GNN4CD_model_88210038325716
// MI455X (gfx1250) — compile-verified, matrix operand order checked by reading, not run on hardware
//
#include <hip/hip_runtime.h>
#include <hip/hip_bf16.h>

// ---------------------------------------------------------------------------
// Model dims
// ---------------------------------------------------------------------------
#define N_LOW   10000
#define N_HIGH  100000
#define SEQ_T   25
#define H_IN    25
#define ENC_D   128
#define L2H_D   64
#define HO_D    64
#define E_LH    400000LL
#define E_HH    1600000LL
#define E_TOT   (E_HH + (long long)N_HIGH)   // hh edges + self loops

// ---------------------------------------------------------------------------
// WMMA helpers (gfx1250 wave32, v_wmma_f32_16x16x32_bf16)
// ---------------------------------------------------------------------------
typedef __attribute__((ext_vector_type(16))) __bf16 bf16x16;
typedef __attribute__((ext_vector_type(8)))  float  f32x8;

__device__ __forceinline__ unsigned pack_bf16(float a, float b) {
  unsigned ua = __float_as_uint(a), ub = __float_as_uint(b);
  ua = (ua + 0x7FFFu + ((ua >> 16) & 1u)) >> 16;   // RNE fp32 -> bf16
  ub = (ub + 0x7FFFu + ((ub >> 16) & 1u)) >> 16;
  return (ua & 0xFFFFu) | (ub << 16);
}

union bfrag { bf16x16 v; unsigned u[8]; };

// A fragment: 16x32 bf16 tile of one 16-row stripe; arow = A + row*K.
// CLAMP=false: branchless unconditional loads (main K loop).
// CLAMP=true : clamp address + select-zero (compile-time K tail only).
template<bool CLAMP>
__device__ __forceinline__ bf16x16 load_a(const float* __restrict__ arow,
                                          int k0, int Klim, int lane) {
  bfrag f;
  int half = lane >> 4;
#pragma unroll
  for (int g = 0; g < 2; ++g) {
    int kb = k0 + half * 8 + g * 16;
#pragma unroll
    for (int j = 0; j < 4; ++j) {
      int k = kb + 2 * j;
      float x0, x1;
      if (CLAMP) {
        float t0 = arow[(k     < Klim) ? k     : 0];
        float t1 = arow[(k + 1 < Klim) ? k + 1 : 0];
        x0 = (k     < Klim) ? t0 : 0.0f;
        x1 = (k + 1 < Klim) ? t1 : 0.0f;
      } else {
        x0 = arow[k]; x1 = arow[k + 1];
      }
      f.u[g * 4 + j] = pack_bf16(x0, x1);
    }
  }
  return f.v;
}

// B fragment: 32x16 slice of W.  WT=true: W=[N][K]; WT=false: W=[K][N].
// N,K are compile-time constants at every call site (inlined), so the n-guard
// folds away for 16-divisible N and the k-guard exists only in the tail.
template<bool WT, bool KCLAMP>
__device__ __forceinline__ bf16x16 load_b(const float* __restrict__ W, int K, int N,
                                          int k0, int n0, int lane) {
  bfrag f;
  int n = n0 + (lane & 15);
  int half = lane >> 4;
  bool nok = ((N & 15) == 0) ? true : (n < N);
  int nn = nok ? n : 0;
#pragma unroll
  for (int g = 0; g < 2; ++g) {
    int kb = k0 + half * 8 + g * 16;
#pragma unroll
    for (int j = 0; j < 4; ++j) {
      int k = kb + 2 * j;
      float x0, x1;
      if (WT) {
        const float* r = W + (long long)nn * K;
        if (KCLAMP) {
          float t0 = r[(k     < K) ? k     : 0];
          float t1 = r[(k + 1 < K) ? k + 1 : 0];
          x0 = (k     < K) ? t0 : 0.0f;
          x1 = (k + 1 < K) ? t1 : 0.0f;
        } else { x0 = r[k]; x1 = r[k + 1]; }
      } else {
        if (KCLAMP) {
          float t0 = W[(long long)((k     < K) ? k     : 0) * N + nn];
          float t1 = W[(long long)((k + 1 < K) ? k + 1 : 0) * N + nn];
          x0 = (k     < K) ? t0 : 0.0f;
          x1 = (k + 1 < K) ? t1 : 0.0f;
        } else {
          x0 = W[(long long)k * N + nn];
          x1 = W[(long long)(k + 1) * N + nn];
        }
      }
      if (!nok) { x0 = 0.0f; x1 = 0.0f; }
      f.u[g * 4 + j] = pack_bf16(x0, x1);
    }
  }
  return f.v;
}

// C = act(A[Mx K] * B + bias). One wave per 16-row tile. All shapes static.
template<int N, int K, bool WT, int ACT, int LDC>
__global__ void wmma_gemm_kernel(const float* __restrict__ A, const float* __restrict__ W,
                                 const float* __restrict__ bias, float* __restrict__ C) {
  constexpr int NT    = (N + 15) / 16;
  constexpr int KMAIN = (K / 32) * 32;
  int lane = threadIdx.x & 31;
  int row0 = blockIdx.x * 16;
  const float* arow = A + (long long)(row0 + (lane & 15)) * K;

  f32x8 acc[NT];
#pragma unroll
  for (int t = 0; t < NT; ++t)
#pragma unroll
    for (int i = 0; i < 8; ++i) acc[t][i] = 0.0f;

  for (int k0 = 0; k0 < KMAIN; k0 += 32) {
    bf16x16 a = load_a<false>(arow, k0, K, lane);
#pragma unroll
    for (int t = 0; t < NT; ++t) {
      bf16x16 b = load_b<WT, false>(W, K, N, k0, t * 16, lane);
      acc[t] = __builtin_amdgcn_wmma_f32_16x16x32_bf16(
          false, a, false, b, (short)0, acc[t], false, false);
    }
  }
  if constexpr (K % 32 != 0) {
    bf16x16 a = load_a<true>(arow, KMAIN, K, lane);
#pragma unroll
    for (int t = 0; t < NT; ++t) {
      bf16x16 b = load_b<WT, true>(W, K, N, KMAIN, t * 16, lane);
      acc[t] = __builtin_amdgcn_wmma_f32_16x16x32_bf16(
          false, a, false, b, (short)0, acc[t], false, false);
    }
  }

  int ncol = lane & 15;
  int half = lane >> 4;
#pragma unroll
  for (int t = 0; t < NT; ++t) {
    int n = t * 16 + ncol;
    if (((N & 15) == 0) || (n < N)) {
      float bv = bias[n];
#pragma unroll
      for (int i = 0; i < 8; ++i) {
        int r = row0 + half * 8 + i;
        float v = acc[t][i] + bv;
        if (ACT == 1) v = fmaxf(v, 0.0f);
        C[(long long)r * LDC + n] = v;
      }
    }
  }
}

// ---------------------------------------------------------------------------
// GRU recurrence: each block owns 16 batch rows, loops all timesteps in LDS.
// ---------------------------------------------------------------------------
#define GRU_R 16
__global__ void gru_recurrent_kernel(const float* __restrict__ gi,
                                     const float* __restrict__ whh,   // [75][25]
                                     const float* __restrict__ bhh,   // [75]
                                     float* __restrict__ hseq) {
  __shared__ float s_w[75 * 25];
  __shared__ float s_b[75];
  __shared__ float s_h[GRU_R][26];
  __shared__ float s_g[GRU_R][75];
  int tid = threadIdx.x;
  for (int i = tid; i < 75 * 25; i += blockDim.x) s_w[i] = whh[i];
  for (int i = tid; i < 75; i += blockDim.x) s_b[i] = bhh[i];
  for (int i = tid; i < GRU_R * 25; i += blockDim.x) s_h[i / 25][i % 25] = 0.0f;
  int row0 = blockIdx.x * GRU_R;
  __syncthreads();
  for (int t = 0; t < SEQ_T; ++t) {
    for (int i = tid; i < GRU_R * 75; i += blockDim.x) {
      int r = i / 75, g = i % 75;
      const float* w = &s_w[g * 25];
      float s = s_b[g];
#pragma unroll
      for (int j = 0; j < 25; ++j) s += s_h[r][j] * w[j];
      s_g[r][g] = s;
    }
    __syncthreads();
    for (int i = tid; i < GRU_R * 25; i += blockDim.x) {
      int r = i / 25, j = i % 25;
      long long grow = ((long long)(row0 + r) * SEQ_T + t) * 75;
      float ir = gi[grow + j], iz = gi[grow + 25 + j], in = gi[grow + 50 + j];
      float hr = s_g[r][j], hz = s_g[r][25 + j], hn = s_g[r][50 + j];
      float rg = 1.0f / (1.0f + __expf(-(ir + hr)));
      float zg = 1.0f / (1.0f + __expf(-(iz + hz)));
      float ng = tanhf(in + rg * hn);
      float h  = (1.0f - zg) * ng + zg * s_h[r][j];
      s_h[r][j] = h;
      hseq[((long long)(row0 + r) * SEQ_T + t) * 25 + j] = h;
    }
    __syncthreads();
  }
}

// ---------------------------------------------------------------------------
// Utility / scatter kernels
// ---------------------------------------------------------------------------
__global__ void fill_f32_kernel(float* p, float v, long long n) {
  for (long long i = blockIdx.x * (long long)blockDim.x + threadIdx.x; i < n;
       i += (long long)gridDim.x * blockDim.x) p[i] = v;
}
__global__ void fill_u32_kernel(unsigned* p, unsigned v, long long n) {
  for (long long i = blockIdx.x * (long long)blockDim.x + threadIdx.x; i < n;
       i += (long long)gridDim.x * blockDim.x) p[i] = v;
}

__global__ void scatter_add_feat_kernel(const float* __restrict__ feat,
                                        const int* __restrict__ src,
                                        const int* __restrict__ dst,
                                        float* __restrict__ out, long long nE, int C) {
  long long total = nE * C;
  for (long long i = blockIdx.x * (long long)blockDim.x + threadIdx.x; i < total;
       i += (long long)gridDim.x * blockDim.x) {
    long long e = i / C; int c = (int)(i % C);
    atomicAdd(&out[(long long)dst[e] * C + c], feat[(long long)src[e] * C + c]);
  }
}

__global__ void deg_count_kernel(const int* __restrict__ dst, float* __restrict__ deg,
                                 long long nE) {
  for (long long i = blockIdx.x * (long long)blockDim.x + threadIdx.x; i < nE;
       i += (long long)gridDim.x * blockDim.x)
    atomicAdd(&deg[dst[i]], 1.0f);
}

__global__ void div_by_deg_kernel(float* __restrict__ x, const float* __restrict__ deg,
                                  long long n, int C) {
  long long total = n * C;
  for (long long i = blockIdx.x * (long long)blockDim.x + threadIdx.x; i < total;
       i += (long long)gridDim.x * blockDim.x) {
    float d = deg[i / C];
    x[i] = x[i] / fmaxf(d, 1.0f);
  }
}

// x[i][0..63] += x_high[i] @ Wroot.T + broot  (K = 7)
__global__ void add_root_kernel(const float* __restrict__ xh, const float* __restrict__ Wr,
                                const float* __restrict__ br, float* __restrict__ x,
                                long long n) {
  long long total = n * 64;
  for (long long i = blockIdx.x * (long long)blockDim.x + threadIdx.x; i < total;
       i += (long long)gridDim.x * blockDim.x) {
    long long node = i >> 6; int o = (int)(i & 63);
    const float* r = Wr + o * 7;
    const float* h = xh + node * 7;
    float s = br[o];
#pragma unroll
    for (int k = 0; k < 7; ++k) s += h[k] * r[k];
    x[i] += s;
  }
}

// ---------------------------------------------------------------------------
// BatchNorm (training mode, biased var): per-block partials + atomics
// ---------------------------------------------------------------------------
__global__ void bn_reduce_kernel(const float* __restrict__ x, float* __restrict__ sums,
                                 long long n, int C, int nodes_per_block) {
  int c    = threadIdx.x % C;
  int sub  = threadIdx.x / C;
  int subs = blockDim.x / C;
  long long start = (long long)blockIdx.x * nodes_per_block;
  long long end   = start + nodes_per_block; if (end > n) end = n;
  float s = 0.0f, q = 0.0f;
  for (long long i = start + sub; i < end; i += subs) {
    float v = x[i * C + c];
    s += v; q += v * v;
  }
  atomicAdd(&sums[c], s);
  atomicAdd(&sums[C + c], q);
}

__global__ void bn_apply_kernel(float* __restrict__ x, const float* __restrict__ sums,
                                const float* __restrict__ g, const float* __restrict__ b,
                                long long n, int C, int relu) {
  long long total = n * C;
  float inv_n = 1.0f / (float)n;
  for (long long i = blockIdx.x * (long long)blockDim.x + threadIdx.x; i < total;
       i += (long long)gridDim.x * blockDim.x) {
    int c = (int)(i % C);
    float m = sums[c] * inv_n;
    float v = sums[C + c] * inv_n - m * m;
    float y = (x[i] - m) * rsqrtf(v + 1e-5f) * g[c] + b[c];
    if (relu) y = fmaxf(y, 0.0f);
    x[i] = y;
  }
}

// ---------------------------------------------------------------------------
// GATv2 edge kernels — one wave32 per (edge, head), lanes over channels.
// Edge e < E_HH -> (hs[e], hd[e]); else self loop e - E_HH.
// ---------------------------------------------------------------------------
__device__ __forceinline__ unsigned fkey(float f) {
  unsigned u = __float_as_uint(f);
  return (u & 0x80000000u) ? ~u : (u | 0x80000000u);
}
__device__ __forceinline__ float funkey(unsigned u) {
  return (u & 0x80000000u) ? __uint_as_float(u & 0x7FFFFFFFu) : __uint_as_float(~u);
}

template<int HEADS>
__global__ void gat_edge_logit_kernel(const float* __restrict__ xl, const float* __restrict__ xr,
                                      const int* __restrict__ hs, const int* __restrict__ hd,
                                      const float* __restrict__ att,
                                      float* __restrict__ logits, unsigned* __restrict__ mmax) {
  const int C = HO_D;  // 64
  int lane = threadIdx.x & 31;
  long long wid    = blockIdx.x * (long long)(blockDim.x >> 5) + (threadIdx.x >> 5);
  long long wstride = (long long)gridDim.x * (blockDim.x >> 5);
  long long total = E_TOT * HEADS;
  for (long long i = wid; i < total; i += wstride) {
    long long e = i / HEADS; int h = (int)(i % HEADS);
    int s, d;
    if (e < E_HH) { s = hs[e]; d = hd[e]; } else { s = d = (int)(e - E_HH); }
    const float* pl = xl + ((long long)s * HEADS + h) * C;
    const float* pr = xr + ((long long)d * HEADS + h) * C;
    const float* pa = att + h * C;
    float v0 = pl[lane]      + pr[lane];
    float v1 = pl[lane + 32] + pr[lane + 32];
    v0 = ((v0 > 0.0f) ? v0 : 0.2f * v0) * pa[lane];
    v1 = ((v1 > 0.0f) ? v1 : 0.2f * v1) * pa[lane + 32];
    float acc = v0 + v1;
#pragma unroll
    for (int o = 16; o > 0; o >>= 1) acc += __shfl_down(acc, o, 32);
    if (lane == 0) {
      logits[i] = acc;
      atomicMax(&mmax[(long long)d * HEADS + h], fkey(acc));
    }
  }
}

template<int HEADS>
__global__ void gat_edge_exp_kernel(const int* __restrict__ hd,
                                    float* __restrict__ logits,
                                    const unsigned* __restrict__ mmax,
                                    float* __restrict__ denom) {
  long long total = E_TOT * HEADS;
  for (long long i = blockIdx.x * (long long)blockDim.x + threadIdx.x; i < total;
       i += (long long)gridDim.x * blockDim.x) {
    long long e = i / HEADS; int h = (int)(i % HEADS);
    int d = (e < E_HH) ? hd[e] : (int)(e - E_HH);
    float a = __expf(logits[i] - funkey(mmax[(long long)d * HEADS + h]));
    logits[i] = a;
    atomicAdd(&denom[(long long)d * HEADS + h], a);
  }
}

template<int HEADS>
__global__ void gat_edge_agg_kernel(const float* __restrict__ xl,
                                    const int* __restrict__ hs, const int* __restrict__ hd,
                                    const float* __restrict__ a,
                                    const float* __restrict__ denom, float* __restrict__ out) {
  const int C = HO_D;
  int lane = threadIdx.x & 31;
  long long wid    = blockIdx.x * (long long)(blockDim.x >> 5) + (threadIdx.x >> 5);
  long long wstride = (long long)gridDim.x * (blockDim.x >> 5);
  long long total = E_TOT * HEADS;
  for (long long i = wid; i < total; i += wstride) {
    long long e = i / HEADS; int h = (int)(i % HEADS);
    int s, d;
    if (e < E_HH) { s = hs[e]; d = hd[e]; } else { s = d = (int)(e - E_HH); }
    float coef = a[i] / denom[(long long)d * HEADS + h];
    const float* pl = xl + ((long long)s * HEADS + h) * C;
    float* po = out + ((long long)d * HEADS + h) * C;
    atomicAdd(&po[lane],      pl[lane]      * coef);
    atomicAdd(&po[lane + 32], pl[lane + 32] * coef);
  }
}

// out = out / deg(mean, incl self loop) + bias   [+ optional relu]
__global__ void gat_finalize_kernel(float* __restrict__ out, const float* __restrict__ deg,
                                    const float* __restrict__ bias, int HC, int relu) {
  long long total = (long long)N_HIGH * HC;
  for (long long i = blockIdx.x * (long long)blockDim.x + threadIdx.x; i < total;
       i += (long long)gridDim.x * blockDim.x) {
    long long node = i / HC; int j = (int)(i % HC);
    float v = out[i] / (deg[node] + 1.0f) + bias[j];
    if (relu) v = fmaxf(v, 0.0f);
    out[i] = v;
  }
}

// Final predictor layer: [N_HIGH,32] @ W[1,32].T + b -> d_out
__global__ void pred_final_kernel(const float* __restrict__ x, const float* __restrict__ W,
                                  const float* __restrict__ b, float* __restrict__ out) {
  for (long long i = blockIdx.x * (long long)blockDim.x + threadIdx.x; i < N_HIGH;
       i += (long long)gridDim.x * blockDim.x) {
    const float* r = x + i * 32;
    float s = b[0];
#pragma unroll
    for (int k = 0; k < 32; ++k) s += r[k] * W[k];
    out[i] = s;
  }
}

// ---------------------------------------------------------------------------
// Host orchestration
// ---------------------------------------------------------------------------
static inline unsigned nblk(long long total, int bs) {
  long long b = (total + bs - 1) / bs;
  if (b > (1LL << 20)) b = 1LL << 20;
  if (b < 1) b = 1;
  return (unsigned)b;
}
static inline unsigned nblk_wave(long long waves) {   // 8 waves per 256-thread block
  long long b = (waves + 7) / 8;
  if (b > (1LL << 20)) b = 1LL << 20;
  if (b < 1) b = 1;
  return (unsigned)b;
}

extern "C" void kernel_launch(void* const* d_in, const int* in_sizes, int n_in,
                              void* d_out, int out_size, void* d_ws, size_t ws_size,
                              hipStream_t stream) {
  (void)in_sizes; (void)n_in; (void)out_size; (void)ws_size;

  // --- inputs (setup_inputs dict order, params flattened in insertion order)
  const float* x_low  = (const float*)d_in[0];
  const float* x_high = (const float*)d_in[1];
  const int*   lh_ei  = (const int*)d_in[2];   // [2][E_LH]
  const int*   hh_ei  = (const int*)d_in[3];   // [2][E_HH]
  const int* ls = lh_ei;            const int* ld = lh_ei + E_LH;
  const int* hs = hh_ei;            const int* hd = hh_ei + E_HH;

  const float* gru_wih[2] = { (const float*)d_in[4],  (const float*)d_in[8]  };
  const float* gru_whh[2] = { (const float*)d_in[5],  (const float*)d_in[9]  };
  const float* gru_bih[2] = { (const float*)d_in[6],  (const float*)d_in[10] };
  const float* gru_bhh[2] = { (const float*)d_in[7],  (const float*)d_in[11] };
  const float* dense_W = (const float*)d_in[12];  const float* dense_b = (const float*)d_in[13];
  const float* dWrel   = (const float*)d_in[14];  const float* dbrel   = (const float*)d_in[15];
  const float* dWroot  = (const float*)d_in[16];  const float* dbroot  = (const float*)d_in[17];
  const float* bn_g[5], *bn_b[5];
  for (int i = 0; i < 5; ++i) { bn_g[i] = (const float*)d_in[18 + 2 * i];
                                bn_b[i] = (const float*)d_in[19 + 2 * i]; }
  const float *gWl[5], *gbl[5], *gWr[5], *gbr[5], *gatt[5], *gbias[5];
  for (int i = 0; i < 5; ++i) {
    int base = 28 + 6 * i;
    gWl[i]   = (const float*)d_in[base + 0];  gbl[i]  = (const float*)d_in[base + 1];
    gWr[i]   = (const float*)d_in[base + 2];  gbr[i]  = (const float*)d_in[base + 3];
    gatt[i]  = (const float*)d_in[base + 4];  gbias[i]= (const float*)d_in[base + 5];
  }
  const float* pW0 = (const float*)d_in[58]; const float* pb0 = (const float*)d_in[59];
  const float* pW1 = (const float*)d_in[60]; const float* pb1 = (const float*)d_in[61];
  const float* pW2 = (const float*)d_in[62]; const float* pb2 = (const float*)d_in[63];
  float* out = (float*)d_out;

  // --- workspace arena (floats)
  float* ws = (float*)d_ws;
  size_t off = 0;
  auto carve = [&](size_t n) { float* p = ws + off; off += (n + 63) & ~(size_t)63; return p; };
  float* GI    = carve((size_t)N_LOW * SEQ_T * 75);
  float* HSA   = carve((size_t)N_LOW * SEQ_T * 25);
  float* HSB   = carve((size_t)N_LOW * SEQ_T * 25);
  float* ENCB  = carve((size_t)N_LOW * ENC_D);
  float* AGG   = carve((size_t)N_HIGH * ENC_D);
  float* DEGLH = carve((size_t)N_HIGH);
  float* XA    = carve((size_t)N_HIGH * 128);
  float* XB    = carve((size_t)N_HIGH * 128);
  float* XL    = carve((size_t)N_HIGH * 128);
  float* XR    = carve((size_t)N_HIGH * 128);
  float* ELOG  = carve((size_t)E_TOT * 2);
  float* DEN   = carve((size_t)N_HIGH * 2);
  unsigned* MM = (unsigned*)carve((size_t)N_HIGH * 2);
  float* DEGHH = carve((size_t)N_HIGH);
  float* BNS   = carve(256);
  float* P2    = XL;                      // reuse for 32-wide MLP activations

  const int BS = 256;

  // ===== 1) GRU encoder =====================================================
  wmma_gemm_kernel<75, 25, true, 0, 75><<<(N_LOW * SEQ_T) / 16, 32, 0, stream>>>(
      x_low, gru_wih[0], gru_bih[0], GI);
  gru_recurrent_kernel<<<N_LOW / GRU_R, 256, 0, stream>>>(GI, gru_whh[0], gru_bhh[0], HSA);
  wmma_gemm_kernel<75, 25, true, 0, 75><<<(N_LOW * SEQ_T) / 16, 32, 0, stream>>>(
      HSA, gru_wih[1], gru_bih[1], GI);
  gru_recurrent_kernel<<<N_LOW / GRU_R, 256, 0, stream>>>(GI, gru_whh[1], gru_bhh[1], HSB);

  // enc = relu(hseq[10000,625] @ dense_W.T + b)
  wmma_gemm_kernel<128, 625, true, 1, 128><<<N_LOW / 16, 32, 0, stream>>>(
      HSB, dense_W, dense_b, ENCB);

  // ===== 2) bipartite GraphConv (mean) ======================================
  fill_f32_kernel<<<nblk((long long)N_HIGH * ENC_D, BS), BS, 0, stream>>>(AGG, 0.0f, (long long)N_HIGH * ENC_D);
  fill_f32_kernel<<<nblk(N_HIGH, BS), BS, 0, stream>>>(DEGLH, 0.0f, N_HIGH);
  scatter_add_feat_kernel<<<nblk(E_LH * ENC_D, BS), BS, 0, stream>>>(ENCB, ls, ld, AGG, E_LH, ENC_D);
  deg_count_kernel<<<nblk(E_LH, BS), BS, 0, stream>>>(ld, DEGLH, E_LH);
  div_by_deg_kernel<<<nblk((long long)N_HIGH * ENC_D, BS), BS, 0, stream>>>(AGG, DEGLH, N_HIGH, ENC_D);

  // x = agg @ Wrel.T + brel + x_high @ Wroot.T + broot
  wmma_gemm_kernel<64, 128, true, 0, 64><<<N_HIGH / 16, 32, 0, stream>>>(AGG, dWrel, dbrel, XA);
  add_root_kernel<<<nblk((long long)N_HIGH * 64, BS), BS, 0, stream>>>(x_high, dWroot, dbroot, XA, N_HIGH);

  // bn[0] on 64-dim features (no relu)
  fill_f32_kernel<<<1, 256, 0, stream>>>(BNS, 0.0f, 256);
  bn_reduce_kernel<<<(unsigned)((N_HIGH + 511) / 512), 256, 0, stream>>>(XA, BNS, N_HIGH, 64, 512);
  bn_apply_kernel<<<nblk((long long)N_HIGH * 64, BS), BS, 0, stream>>>(XA, BNS, bn_g[0], bn_b[0], N_HIGH, 64, 0);

  // hh degree (self loop added in finalize)
  fill_f32_kernel<<<nblk(N_HIGH, BS), BS, 0, stream>>>(DEGHH, 0.0f, N_HIGH);
  deg_count_kernel<<<nblk(E_HH, BS), BS, 0, stream>>>(hd, DEGHH, E_HH);

  // ===== 3) GATv2 processor =================================================
  float* X = XA;
  float* Y = XB;
  for (int layer = 0; layer < 5; ++layer) {
    const int heads = (layer < 4) ? 2 : 1;
    const int HC    = heads * HO_D;
    const int relu_out = (layer == 4) ? 1 : 0;
    long long EHt = E_TOT * heads;

    // xl / xr projections: x @ Wl + bl   (Wl is [K][N] -> WT=false)
    if (layer == 0) {
      wmma_gemm_kernel<128, 64, false, 0, 128><<<N_HIGH / 16, 32, 0, stream>>>(X, gWl[0], gbl[0], XL);
      wmma_gemm_kernel<128, 64, false, 0, 128><<<N_HIGH / 16, 32, 0, stream>>>(X, gWr[0], gbr[0], XR);
    } else if (layer < 4) {
      wmma_gemm_kernel<128, 128, false, 0, 128><<<N_HIGH / 16, 32, 0, stream>>>(X, gWl[layer], gbl[layer], XL);
      wmma_gemm_kernel<128, 128, false, 0, 128><<<N_HIGH / 16, 32, 0, stream>>>(X, gWr[layer], gbr[layer], XR);
    } else {
      wmma_gemm_kernel<64, 128, false, 0, 64><<<N_HIGH / 16, 32, 0, stream>>>(X, gWl[4], gbl[4], XL);
      wmma_gemm_kernel<64, 128, false, 0, 64><<<N_HIGH / 16, 32, 0, stream>>>(X, gWr[4], gbr[4], XR);
    }

    fill_u32_kernel<<<nblk((long long)N_HIGH * heads, BS), BS, 0, stream>>>(MM, 0u, (long long)N_HIGH * heads);
    fill_f32_kernel<<<nblk((long long)N_HIGH * heads, BS), BS, 0, stream>>>(DEN, 0.0f, (long long)N_HIGH * heads);
    fill_f32_kernel<<<nblk((long long)N_HIGH * HC, BS), BS, 0, stream>>>(Y, 0.0f, (long long)N_HIGH * HC);

    if (heads == 2) {
      gat_edge_logit_kernel<2><<<nblk_wave(EHt), BS, 0, stream>>>(XL, XR, hs, hd, gatt[layer], ELOG, MM);
      gat_edge_exp_kernel<2><<<nblk(EHt, BS), BS, 0, stream>>>(hd, ELOG, MM, DEN);
      gat_edge_agg_kernel<2><<<nblk_wave(EHt), BS, 0, stream>>>(XL, hs, hd, ELOG, DEN, Y);
    } else {
      gat_edge_logit_kernel<1><<<nblk_wave(EHt), BS, 0, stream>>>(XL, XR, hs, hd, gatt[layer], ELOG, MM);
      gat_edge_exp_kernel<1><<<nblk(EHt, BS), BS, 0, stream>>>(hd, ELOG, MM, DEN);
      gat_edge_agg_kernel<1><<<nblk_wave(EHt), BS, 0, stream>>>(XL, hs, hd, ELOG, DEN, Y);
    }
    gat_finalize_kernel<<<nblk((long long)N_HIGH * HC, BS), BS, 0, stream>>>(Y, DEGHH, gbias[layer], HC, relu_out);

    if (layer < 4) {
      fill_f32_kernel<<<1, 256, 0, stream>>>(BNS, 0.0f, 256);
      bn_reduce_kernel<<<(unsigned)((N_HIGH + 511) / 512), 256, 0, stream>>>(Y, BNS, N_HIGH, 128, 512);
      bn_apply_kernel<<<nblk((long long)N_HIGH * 128, BS), BS, 0, stream>>>(Y, BNS, bn_g[layer + 1], bn_b[layer + 1], N_HIGH, 128, 1);
    }
    float* tmp = X; X = Y; Y = tmp;
  }
  // X now holds relu(gat4) [N_HIGH, 64]

  // ===== 4) predictor MLP ===================================================
  wmma_gemm_kernel<64, 64, true, 1, 64><<<N_HIGH / 16, 32, 0, stream>>>(X, pW0, pb0, Y);
  wmma_gemm_kernel<32, 64, true, 1, 32><<<N_HIGH / 16, 32, 0, stream>>>(Y, pW1, pb1, P2);
  pred_final_kernel<<<nblk(N_HIGH, BS), BS, 0, stream>>>(P2, pW2, pb2, out);
}
